// LightGCNEncoder_15994458210392
// MI455X (gfx1250) — matrix operand order repaired and checked by reading, not compiled, not run on hardware
//
#include <hip/hip_runtime.h>
#include <hip/hip_bf16.h>
#include <stdint.h>

// Problem constants (from reference)
#define NND  100000
#define NED  1600000
#define HID  128
#define OUTD 64

typedef __attribute__((ext_vector_type(16))) __bf16 v16bf;
typedef __attribute__((ext_vector_type(8)))  __bf16 v8bf;
typedef __attribute__((ext_vector_type(8)))  float  v8f;

__device__ __forceinline__ void atomic_add_f32(float* p, float v) {
    __hip_atomic_fetch_add(p, v, __ATOMIC_RELAXED, __HIP_MEMORY_SCOPE_AGENT);
}

__global__ void k_zero_u32(unsigned int* p, int n) {
    int i = blockIdx.x * blockDim.x + threadIdx.x;
    if (i < n) p[i] = 0u;
}

__global__ void k_zero_f32v4(float4* p, int n4) {
    int i = blockIdx.x * blockDim.x + threadIdx.x;
    if (i < n4) p[i] = make_float4(0.f, 0.f, 0.f, 0.f);
}

__global__ void k_degree(const int* __restrict__ rows, unsigned int* __restrict__ deg, int E) {
    int e = blockIdx.x * blockDim.x + threadIdx.x;
    if (e < E) atomicAdd(&deg[rows[e]], 1u);
}

__global__ void k_deg_inv_sqrt(const unsigned int* __restrict__ deg, float* __restrict__ dis, int n) {
    int i = blockIdx.x * blockDim.x + threadIdx.x;
    if (i < n) {
        float d = (float)deg[i];
        dis[i] = rsqrtf(d < 1.0f ? 1.0f : d);
    }
}

__global__ void k_edge_norm(const int* __restrict__ rows, const int* __restrict__ cols,
                            const float* __restrict__ dis, float* __restrict__ nrm, int E) {
    int e = blockIdx.x * blockDim.x + threadIdx.x;
    if (e < E) nrm[e] = dis[rows[e]] * dis[cols[e]];
}

__global__ void k_init(const float4* __restrict__ x, float4* __restrict__ hA,
                       float4* __restrict__ acc, int n4) {
    int i = blockIdx.x * blockDim.x + threadIdx.x;
    if (i < n4) { float4 v = x[i]; hA[i] = v; acc[i] = v; }
}

// One wave (32 lanes) per edge; each lane moves 4 contiguous floats (128 total).
__global__ void k_scatter(const float4* __restrict__ hsrc, float* __restrict__ hdst,
                          const int* __restrict__ rows, const int* __restrict__ cols,
                          const float* __restrict__ nrm, int E) {
    int t    = blockIdx.x * blockDim.x + threadIdx.x;
    int e    = t >> 5;
    int lane = t & 31;
    if (e >= E) return;
    int   r = rows[e];
    int   c = cols[e];
    float w = nrm[e];
    float4 v = hsrc[(size_t)c * (HID / 4) + lane];
    float* d = hdst + (size_t)r * HID + lane * 4;
    atomic_add_f32(d + 0, v.x * w);
    atomic_add_f32(d + 1, v.y * w);
    atomic_add_f32(d + 2, v.z * w);
    atomic_add_f32(d + 3, v.w * w);
}

__global__ void k_accum(float4* __restrict__ acc, const float4* __restrict__ h, int n4) {
    int i = blockIdx.x * blockDim.x + threadIdx.x;
    if (i < n4) {
        float4 a = acc[i], b = h[i];
        a.x += b.x; a.y += b.y; a.z += b.z; a.w += b.w;
        acc[i] = a;
    }
}

__global__ void k_cvt_bf16(const float* __restrict__ a, __bf16* __restrict__ o, int n, float scale) {
    int i = blockIdx.x * blockDim.x + threadIdx.x;
    if (i < n) o[i] = (__bf16)(a[i] * scale);
}

// out[N,64] = A[N,128](bf16) @ Wb[64,128]^T (bf16) + bias.
// One wave computes one 16x16 output tile via 4x v_wmma_f32_16x16x32_bf16.
__global__ void k_gemm_wmma(const __bf16* __restrict__ A, const __bf16* __restrict__ Wb,
                            const float* __restrict__ bias, float* __restrict__ out, int N) {
    const int tilesN = OUTD / 16;                       // 4
    int gwave = blockIdx.x * (blockDim.x >> 5) + (threadIdx.x >> 5);
    int lane  = threadIdx.x & 31;
    int tileM = gwave / tilesN;
    int tileN = gwave - tileM * tilesN;
    if (tileM * 16 >= N) return;                        // wave-uniform

    int half = lane >> 4;                               // 0: lanes 0-15, 1: lanes 16-31
    int l15  = lane & 15;
    int rowA = tileM * 16 + l15;                        // A: M = lane%16
    int colN = tileN * 16 + l15;                        // B: N = lane%16
    const __bf16* Arow = A  + (size_t)rowA * HID;
    const __bf16* Wrow = Wb + (size_t)colN * HID;       // B[k][n] = W[n][k]

    v8f c = {};
#pragma unroll
    for (int k0 = 0; k0 < HID; k0 += 32) {
        union { v16bf v; v8bf h[2]; } a, b;
        // 16-bit A-matrix 16x32 layout: lanes0-15 hold K 0-7 then 16-23; lanes16-31 K 8-15 then 24-31
        a.h[0] = *(const v8bf*)(Arow + k0 + half * 8);
        a.h[1] = *(const v8bf*)(Arow + k0 + 16 + half * 8);
        // B-matrix 32x16: lanes0-15 hold K 0-7 then 16-23; lanes16-31 K 8-15 then 24-31 for column N=lane%16
        b.h[0] = *(const v8bf*)(Wrow + k0 + half * 8);
        b.h[1] = *(const v8bf*)(Wrow + k0 + 16 + half * 8);
        c = __builtin_amdgcn_wmma_f32_16x16x32_bf16(false, a.v, false, b.v,
                                                    (short)0, c, false, false);
    }

    float bc = bias[colN];
#pragma unroll
    for (int i = 0; i < 8; i++) {
        int m = tileM * 16 + i + half * 8;              // C/D: VGPR i -> M=i (lanes0-15) / M=i+8
        out[(size_t)m * OUTD + colN] = c[i] + bc;
    }
}

extern "C" void kernel_launch(void* const* d_in, const int* in_sizes, int n_in,
                              void* d_out, int out_size, void* d_ws, size_t ws_size,
                              hipStream_t stream) {
    const float* x    = (const float*)d_in[0];
    const int*   ei   = (const int*)d_in[1];   // [2, E] int32 (JAX x64-off downcasts int64)
    const float* W    = (const float*)d_in[2]; // [64, 128]
    const float* bias = (const float*)d_in[3]; // [64]
    float*       out  = (float*)d_out;

    const int* rows = ei;              // destination nodes (segment ids)
    const int* cols = ei + NED;        // source nodes

    // Workspace carve-up (256B aligned)
    char* ws = (char*)d_ws;
    auto al = [](size_t v) { return (v + 255) & ~(size_t)255; };
    size_t off = 0;
    unsigned int* deg = (unsigned int*)(ws + off); off = al(off + (size_t)NND * 4);
    float* dis = (float*)(ws + off);               off = al(off + (size_t)NND * 4);
    float* nrm = (float*)(ws + off);               off = al(off + (size_t)NED * 4);
    size_t offhA = off;
    float* hA  = (float*)(ws + off);               off = al(off + (size_t)NND * HID * 4);
    float* hB  = (float*)(ws + off);               off = al(off + (size_t)NND * HID * 4);
    float* acc = (float*)(ws + off);               off = al(off + (size_t)NND * HID * 4);
    __bf16* Wbf = (__bf16*)(ws + off);             off = al(off + (size_t)OUTD * HID * 2);
    __bf16* accbf = (__bf16*)(ws + offhA);         // aliases hA (dead by conversion time)

    const int NH  = NND * HID;         // 12.8M floats
    const int NH4 = NH / 4;

    k_zero_u32<<<(NND + 255) / 256, 256, 0, stream>>>(deg, NND);
    k_degree<<<(NED + 255) / 256, 256, 0, stream>>>(rows, deg, NED);
    k_deg_inv_sqrt<<<(NND + 255) / 256, 256, 0, stream>>>(deg, dis, NND);
    k_edge_norm<<<(NED + 255) / 256, 256, 0, stream>>>(rows, cols, dis, nrm, NED);
    k_init<<<(NH4 + 255) / 256, 256, 0, stream>>>((const float4*)x, (float4*)hA, (float4*)acc, NH4);

    float* src = hA;
    float* dst = hB;
    for (int l = 0; l < 3; ++l) {
        k_zero_f32v4<<<(NH4 + 255) / 256, 256, 0, stream>>>((float4*)dst, NH4);
        long long sthreads = (long long)NED * 32;              // wave per edge
        k_scatter<<<(int)((sthreads + 255) / 256), 256, 0, stream>>>(
            (const float4*)src, dst, rows, cols, nrm, NED);
        k_accum<<<(NH4 + 255) / 256, 256, 0, stream>>>((float4*)acc, (const float4*)dst, NH4);
        float* t = src; src = dst; dst = t;
    }

    // mean over {x, h1, h2, h3} folded into the bf16 conversion (scale 0.25)
    k_cvt_bf16<<<(NH + 255) / 256, 256, 0, stream>>>(acc, accbf, NH, 0.25f);
    k_cvt_bf16<<<(OUTD * HID + 255) / 256, 256, 0, stream>>>(W, Wbf, OUTD * HID, 1.0f);

    // (100000/16) * (64/16) = 25000 tiles; 4 waves (128 threads) per block -> 6250 blocks exact
    k_gemm_wmma<<<(NND / 16) * (OUTD / 16) / 4, 128, 0, stream>>>(accbf, Wbf, bias, out, NND);
}
